// SumLayer_68616397521082
// MI455X (gfx1250) — matrix operand order repaired and checked by reading, not compiled, not run on hardware
//
#include <hip/hip_runtime.h>
#include <math.h>

// Problem dims (fixed by the reference)
#define B_  256
#define R_  512
#define KO_ 32
#define KI_ 32

typedef _Float16 v16h __attribute__((ext_vector_type(16)));
typedef _Float16 v8h  __attribute__((ext_vector_type(8)));
typedef float    v8f  __attribute__((ext_vector_type(8)));

// ---------------------------------------------------------------------------
// Phase 1: p[r,ko,:] = softmax_ki(weight[r,ko,:]) stored as f16 (natural
// (R,KO,KI) layout; KI contiguous so phase-2 B-fragment loads are b128).
// One thread per (r,ko) row: 32 f32 in, 32 f16 out. R*KO = 16384 threads.
// ---------------------------------------------------------------------------
__global__ void wsoftmax_kernel(const float* __restrict__ w,
                                _Float16* __restrict__ ew) {
    const int row = blockIdx.x * blockDim.x + threadIdx.x;   // exact: no guard
    const float* wp = w + (size_t)row * KI_;
    float v[KI_];
#pragma unroll
    for (int i = 0; i < KI_ / 4; ++i) {
        float4 t = ((const float4*)wp)[i];
        v[4 * i + 0] = t.x; v[4 * i + 1] = t.y;
        v[4 * i + 2] = t.z; v[4 * i + 3] = t.w;
    }
    float mx = v[0];
#pragma unroll
    for (int i = 1; i < KI_; ++i) mx = fmaxf(mx, v[i]);
    float s = 0.0f;
#pragma unroll
    for (int i = 0; i < KI_; ++i) { v[i] = __expf(v[i] - mx); s += v[i]; }
    const float inv = 1.0f / s;
    v8h o[4];
#pragma unroll
    for (int i = 0; i < KI_; ++i) o[i / 8][i % 8] = (_Float16)(v[i] * inv);
    v8h* op = (v8h*)(ew + (size_t)row * KI_);
#pragma unroll
    for (int i = 0; i < 4; ++i) op[i] = o[i];
}

// ---------------------------------------------------------------------------
// Phase 2: one wave32 per (16-batch tile, region r).
//   A (16x32 f16)  = exp(x[btile, r, :] - rowmax)       -> 1 fragment
//   B (32x16 f16)  = p[r, ko_tile, :]^T                 -> 2 fragments (KO=32)
//   D = A x B (f32 16x16)  via v_wmma_f32_16x16x32_f16
//   y = rowmax + log(D)
//
// Fragment layouts used here (contiguous k order, same for A and B):
//   A: lane L holds row M=L%16; elems 0..15 = K 16h..16h+15
//      (h = L/16)  -> two contiguous 8-float runs of x.
//   B: lane L holds col N=L%16; elems 0..15 = K 16h..16h+15
//      -> 16 contiguous f16 of one p row (two b128 loads).
//   D: lane L, VGPR v -> (M = v + 8h, N = L%16).
// ---------------------------------------------------------------------------
__global__ void lse_mm_kernel(const float* __restrict__ x,
                              const _Float16* __restrict__ ew,
                              float* __restrict__ out) {
    const int lane = threadIdx.x & 31;
    const int wave = threadIdx.x >> 5;
    const int gw   = blockIdx.x * 8 + wave;   // 0 .. 16*R_-1, exact
    const int bt   = gw & 15;                 // batch tile (16 rows)
    const int r    = gw >> 4;                 // region
    const int half = lane >> 4;
    const int m    = lane & 15;
    const int b    = bt * 16 + m;

    // ---- A fragment: two 32B runs of x, row max, exp, cvt f16 ----
    const float* xp = x + ((size_t)b * R_ + r) * KI_;
    float xv[16];
    {
        const float4* p0 = (const float4*)(xp + 16 * half);       // K 16h..16h+7
        const float4* p1 = (const float4*)(xp + 16 * half + 8);   // K 16h+8..
        float4 a0 = p0[0], a1 = p0[1], a2 = p1[0], a3 = p1[1];
        xv[0] = a0.x;  xv[1] = a0.y;  xv[2]  = a0.z;  xv[3]  = a0.w;
        xv[4] = a1.x;  xv[5] = a1.y;  xv[6]  = a1.z;  xv[7]  = a1.w;
        xv[8] = a2.x;  xv[9] = a2.y;  xv[10] = a2.z;  xv[11] = a2.w;
        xv[12] = a3.x; xv[13] = a3.y; xv[14] = a3.z;  xv[15] = a3.w;
    }
    float mx = xv[0];
#pragma unroll
    for (int i = 1; i < 16; ++i) mx = fmaxf(mx, xv[i]);
    // lanes m and m+16 hold halves of row m: combine across the half-waves
    mx = fmaxf(mx, __shfl_xor(mx, 16, 32));   // wave32

    v16h afrag;
#pragma unroll
    for (int e = 0; e < 16; ++e) afrag[e] = (_Float16)__expf(xv[e] - mx);

    // ---- B fragments: p rows are contiguous f16; elem e <-> K = 16h+e ----
    v16h bfrag[2];
#pragma unroll
    for (int t = 0; t < 2; ++t) {
        const _Float16* ep =
            ew + ((size_t)r * KO_ + (t * 16 + m)) * KI_ + 16 * half;
        v8h lo = ((const v8h*)ep)[0];
        v8h hi = ((const v8h*)ep)[1];
#pragma unroll
        for (int e = 0; e < 8; ++e) { bfrag[t][e] = lo[e]; bfrag[t][e + 8] = hi[e]; }
    }

    // ---- matrix cores: full K=32 reduction in one op per KO tile ----
    v8f c0 = {}, c1 = {};
    c0 = __builtin_amdgcn_wmma_f32_16x16x32_f16(false, afrag, false, bfrag[0],
                                                (short)0, c0, false, false);
    c1 = __builtin_amdgcn_wmma_f32_16x16x32_f16(false, afrag, false, bfrag[1],
                                                (short)0, c1, false, false);

    // ---- y = rowmax + log(S); D VGPR v -> row v+8h, col m ----
#pragma unroll
    for (int v = 0; v < 8; ++v) {
        const int row   = v + 8 * half;
        const float mr  = __shfl(mx, row, 32);      // lane 'row' owns that max
        const float y0  = mr + __logf(c0[v]);
        const float y1  = mr + __logf(c1[v]);
        float* op = out + ((size_t)(bt * 16 + row) * R_ + r) * KO_;
        op[m]      = y0;   // lanes 0..15 contiguous in ko -> 64B segments
        op[16 + m] = y1;
    }
}

// ---------------------------------------------------------------------------
extern "C" void kernel_launch(void* const* d_in, const int* in_sizes, int n_in,
                              void* d_out, int out_size, void* d_ws,
                              size_t ws_size, hipStream_t stream) {
    const float* x = (const float*)d_in[0];   // (B, R, KI) f32
    const float* w = (const float*)d_in[1];   // (R, KO, KI) f32
    float* out     = (float*)d_out;           // (B, R, KO) f32
    _Float16* ew   = (_Float16*)d_ws;         // (R, KO, KI) f16 = 1 MB scratch

    // Phase 1: softmax(weight) -> f16 workspace. 16384 rows, 256 thr/block.
    wsoftmax_kernel<<<(R_ * KO_) / 256, 256, 0, stream>>>(w, ew);

    // Phase 2: 16 batch-tiles * 512 regions = 8192 waves, 8 waves/block.
    lse_mm_kernel<<<(16 * R_) / 8, 256, 0, stream>>>(x, ew, out);
}
